// MultiHeadAttention_51384988729734
// MI455X (gfx1250) — matrix operand order repaired and checked by reading, not compiled, not run on hardware
//
#include <hip/hip_runtime.h>
#include <hip/hip_bf16.h>

// ---------------------------------------------------------------------------
// MHA for MI455X (gfx1250, wave32). All matmuls via v_wmma_f32_16x16x32_f16.
// f16->LDS staging uses GLOBAL_LOAD_ASYNC_TO_LDS_B128 (ASYNCcnt path).
// ---------------------------------------------------------------------------

typedef __attribute__((ext_vector_type(16))) _Float16 v16h;
typedef __attribute__((ext_vector_type(8)))  float    v8f;

#define DIMSZ   1024
#define HEADS   16
#define HDIM    64
#define BATCH   2
#define SEQ     2048
#define MROWS   (BATCH * SEQ)          // 4096
#define ATT_SCALE (0.03125f)           // 1/sqrt(1024)

#define TM 128
#define TN 128
#define TK 32
#define LDA (TK + 4)                   // LDS row pad (f16 elems)

// ---- CDNA5 async copy helpers --------------------------------------------

// GLOBAL_LOAD_ASYNC_TO_LDS_B128: VDST = LDS byte offset (VGPR), VADDR = 64-bit
// global address. Tracked with ASYNCcnt (ISA 15.18.3 op 98).
__device__ __forceinline__ void async_copy16(const void* gptr, void* lptr) {
  unsigned lds_off = (unsigned)(size_t)lptr;          // addr[31:0] == LDS offset
  unsigned long long ga = (unsigned long long)(size_t)gptr;
  asm volatile("global_load_async_to_lds_b128 %0, %1, off"
               :: "v"(lds_off), "v"(ga) : "memory");
}
__device__ __forceinline__ void async_wait0() {
  asm volatile("s_wait_asynccnt 0x0" ::: "memory");
}

// ---- WMMA helpers ---------------------------------------------------------

__device__ __forceinline__ v8f wmma16(v16h a, v16h b, v8f c) {
  // D = A(16x32 f16) x B(32x16 f16) + C(16x16 f32)
  return __builtin_amdgcn_wmma_f32_16x16x32_f16(
      /*neg_a=*/false, a, /*neg_b=*/false, b,
      /*c_mod=*/(short)0, c, /*reuse_a=*/false, /*reuse_b=*/false);
}

// A fragment K index (contiguous: lanes 0-15 K=0..15, lanes 16-31 K=16..31)
__device__ __forceinline__ int afrag_k(int p, int lane) {
  return ((lane >> 4) & 1) * 16 + p * 2;
}
// B fragment K index (contiguous, same as A: lanes 0-15 K=0..15, lanes 16-31 K=16..31)
__device__ __forceinline__ int bfrag_k(int p, int lane) {
  return ((lane >> 4) & 1) * 16 + p * 2;
}

// Load A fragment from row-major f16 [16 rows x >=32 cols], stride in elems.
__device__ __forceinline__ v16h load_afrag(const _Float16* base, int stride, int lane) {
  const _Float16* r = base + (size_t)(lane & 15) * stride;
  v16h a;
#pragma unroll
  for (int p = 0; p < 8; ++p) {
    int k = afrag_k(p, lane);
    a[2 * p]     = r[k];
    a[2 * p + 1] = r[k + 1];
  }
  return a;
}

// Load B fragment where source rows are the N dimension and K is contiguous
// (i.e. W stored [N][K] and we need B[k][n] = W[n][k]).
__device__ __forceinline__ v16h load_bfrag_t(const _Float16* base, int stride, int lane) {
  const _Float16* r = base + (size_t)(lane & 15) * stride;
  v16h b;
#pragma unroll
  for (int p = 0; p < 8; ++p) {
    int k = bfrag_k(p, lane);
    b[2 * p]     = r[k];
    b[2 * p + 1] = r[k + 1];
  }
  return b;
}

// Load B fragment from natural [K rows][N cols] layout, stride in elems.
__device__ __forceinline__ v16h load_bfrag_n(const _Float16* base, int stride, int lane) {
  int col = lane & 15;
  v16h b;
#pragma unroll
  for (int p = 0; p < 8; ++p) {
    int k = bfrag_k(p, lane);
    b[2 * p]     = base[(size_t)k * stride + col];
    b[2 * p + 1] = base[(size_t)(k + 1) * stride + col];
  }
  return b;
}

// ---------------------------------------------------------------------------
// Kernel 1: projection GEMM  Out = X @ W^T + bias, stored f16 in [B,H,S,64].
// X: [M=4096, K=1024] fp32 ; W: [N=1024, K=1024] fp32 ; bias: [N] fp32.
// (fp32 inputs need in-flight f16 conversion -> VGPR staging, not async.)
// ---------------------------------------------------------------------------
__global__ __launch_bounds__(256, 2)
void mha_proj_gemm(const float* __restrict__ X, const float* __restrict__ W,
                   const float* __restrict__ bias, _Float16* __restrict__ Out) {
  __shared__ _Float16 As[TM][LDA];
  __shared__ _Float16 Ws[TN][LDA];

  const int t = threadIdx.x;
  const int lane = t & 31;
  const int wave = t >> 5;          // 0..7
  const int wm = wave & 3;          // 4 row-groups of 32
  const int wn = wave >> 2;         // 2 col-groups of 64
  const int m0 = blockIdx.y * TM;
  const int n0 = blockIdx.x * TN;

  v8f acc[2][4];
#pragma unroll
  for (int i = 0; i < 2; ++i)
#pragma unroll
    for (int j = 0; j < 4; ++j) { v8f z = {}; acc[i][j] = z; }

  for (int k0 = 0; k0 < DIMSZ; k0 += TK) {
    // stage A and W tiles (fp32 -> f16)
#pragma unroll
    for (int i = t; i < TM * 8; i += 256) {
      int r = i >> 3, c = (i & 7) * 4;
      float4 fa = *(const float4*)(X + (size_t)(m0 + r) * DIMSZ + k0 + c);
      As[r][c + 0] = (_Float16)fa.x; As[r][c + 1] = (_Float16)fa.y;
      As[r][c + 2] = (_Float16)fa.z; As[r][c + 3] = (_Float16)fa.w;
      float4 fw = *(const float4*)(W + (size_t)(n0 + r) * DIMSZ + k0 + c);
      Ws[r][c + 0] = (_Float16)fw.x; Ws[r][c + 1] = (_Float16)fw.y;
      Ws[r][c + 2] = (_Float16)fw.z; Ws[r][c + 3] = (_Float16)fw.w;
    }
    if (k0 + TK < DIMSZ)
      __builtin_prefetch(X + (size_t)(m0 + (t >> 1)) * DIMSZ + k0 + TK, 0, 1);
    __syncthreads();

    v16h af[2], bf[4];
#pragma unroll
    for (int fm = 0; fm < 2; ++fm)
      af[fm] = load_afrag(&As[wm * 32 + fm * 16][0], LDA, lane);
#pragma unroll
    for (int fn = 0; fn < 4; ++fn)
      bf[fn] = load_bfrag_t(&Ws[wn * 64 + fn * 16][0], LDA, lane);
#pragma unroll
    for (int fm = 0; fm < 2; ++fm)
#pragma unroll
      for (int fn = 0; fn < 4; ++fn)
        acc[fm][fn] = wmma16(af[fm], bf[fn], acc[fm][fn]);
    __syncthreads();
  }

  // store f16 in head layout [B,H,S,64]
  const int colc = lane & 15;
  const int half8 = ((lane >> 4) & 1) * 8;
#pragma unroll
  for (int fm = 0; fm < 2; ++fm)
#pragma unroll
    for (int fn = 0; fn < 4; ++fn)
#pragma unroll
      for (int v = 0; v < 8; ++v) {
        int m = m0 + wm * 32 + fm * 16 + v + half8;
        int n = n0 + wn * 64 + fn * 16 + colc;
        float val = acc[fm][fn][v] + bias[n];
        int b = m >> 11;            // m / SEQ
        int s = m & (SEQ - 1);
        int h = n >> 6;
        int d = n & 63;
        Out[(((size_t)b * HEADS + h) * SEQ + s) * HDIM + d] = (_Float16)val;
      }
}

// ---------------------------------------------------------------------------
// Kernel 2: flash attention. 1 wave = 16 query rows; 4 waves/block share K/V
// tiles in LDS (staged with async-to-LDS copies). Qp/Kp/Vp: f16 [B,H,S,64].
// mask: int [B,1,Sq,Skv]. Y: f16 [B,Sq,1024], col = h*64+d.
// ---------------------------------------------------------------------------
__global__ __launch_bounds__(128, 2)
void mha_attn(const _Float16* __restrict__ Qp, const _Float16* __restrict__ Kp,
              const _Float16* __restrict__ Vp, const int* __restrict__ mask,
              _Float16* __restrict__ Y) {
  __shared__ _Float16 Kt[32][HDIM + 8];     // row pad keeps 16B alignment
  __shared__ _Float16 Vt[32][HDIM + 8];
  __shared__ _Float16 Pb[4][16][36];

  const int t = threadIdx.x;
  const int lane = t & 31;
  const int wave = t >> 5;                  // 0..3
  const int bh = blockIdx.y;                // b*H + h
  const int b = bh / HEADS;
  const int h = bh % HEADS;
  const int q0 = blockIdx.x * 64 + wave * 16;

  // preload Q A-fragments (dk=64 -> two K=32 chunks), rows q0..q0+15
  const _Float16* qbase =
      Qp + ((size_t)bh * SEQ + q0 + (lane & 15)) * HDIM;
  v16h qf[2];
#pragma unroll
  for (int c = 0; c < 2; ++c)
#pragma unroll
    for (int p = 0; p < 8; ++p) {
      int k = afrag_k(p, lane);
      qf[c][2 * p]     = qbase[c * 32 + k];
      qf[c][2 * p + 1] = qbase[c * 32 + k + 1];
    }

  float m_r[8], l_r[8];
#pragma unroll
  for (int v = 0; v < 8; ++v) { m_r[v] = -1.0e30f; l_r[v] = 0.0f; }
  v8f O[4];
#pragma unroll
  for (int d = 0; d < 4; ++d) { v8f z = {}; O[d] = z; }

  const int colc = lane & 15;
  const int half8 = ((lane >> 4) & 1) * 8;
  const int* mbase = mask + (size_t)b * SEQ * SEQ;

  for (int j0 = 0; j0 < SEQ; j0 += 32) {
    __syncthreads();
    // async stage of 32x64 K and V tiles: 16B per copy, no VGPR round-trip
    size_t kvbase = ((size_t)bh * SEQ + j0) * HDIM;
#pragma unroll
    for (int i = t; i < 32 * 8; i += 128) {
      int r = i >> 3, c8 = (i & 7) * 8;
      async_copy16(Kp + kvbase + (size_t)r * HDIM + c8, &Kt[r][c8]);
      async_copy16(Vp + kvbase + (size_t)r * HDIM + c8, &Vt[r][c8]);
    }
    async_wait0();
    __syncthreads();

    // S = Q . K^T for a 16x32 score strip (two 16x16 fragments)
    v8f sc[2];
#pragma unroll
    for (int n = 0; n < 2; ++n) {
      v8f s = {};
#pragma unroll
      for (int c = 0; c < 2; ++c) {
        // B[k][col] = Kt[n*16+col][c*32+k]  (K rows are kv positions)
        v16h bK = load_bfrag_t(&Kt[n * 16][c * 32], HDIM + 8, lane);
        s = wmma16(qf[c], bK, s);
      }
      sc[n] = s;
    }

    // scale + mask + online softmax
#pragma unroll
    for (int v = 0; v < 8; ++v) {
      int qrow = q0 + v + half8;
      const int* mq = mbase + (size_t)qrow * SEQ + j0;
      float x0 = sc[0][v] * ATT_SCALE;
      float x1 = sc[1][v] * ATT_SCALE;
      if (mq[colc] != 0)       x0 = -1.0e30f;
      if (mq[16 + colc] != 0)  x1 = -1.0e30f;

      float tmax = fmaxf(x0, x1);
#pragma unroll
      for (int off = 8; off >= 1; off >>= 1)
        tmax = fmaxf(tmax, __shfl_xor(tmax, off, 32));
      float mnew = fmaxf(m_r[v], tmax);
      float alpha = __expf(m_r[v] - mnew);
      m_r[v] = mnew;
      float p0 = __expf(x0 - mnew);
      float p1 = __expf(x1 - mnew);
      float rs = p0 + p1;
#pragma unroll
      for (int off = 8; off >= 1; off >>= 1)
        rs += __shfl_xor(rs, off, 32);
      l_r[v] = l_r[v] * alpha + rs;
#pragma unroll
      for (int d = 0; d < 4; ++d) O[d][v] *= alpha;
      Pb[wave][v + half8][colc]      = (_Float16)p0;
      Pb[wave][v + half8][16 + colc] = (_Float16)p1;
    }
    __syncthreads();

    // O += P(16x32) . V(32x64)
    v16h pa = load_afrag(&Pb[wave][0][0], 36, lane);
#pragma unroll
    for (int d = 0; d < 4; ++d) {
      v16h bV = load_bfrag_n(&Vt[0][d * 16], HDIM + 8, lane);
      O[d] = wmma16(pa, bV, O[d]);
    }
  }

  // normalize and store f16 [B,Sq,1024]
#pragma unroll
  for (int v = 0; v < 8; ++v) {
    float inv = 1.0f / l_r[v];
    int qrow = q0 + v + half8;
    _Float16* yrow = Y + ((size_t)b * SEQ + qrow) * DIMSZ + h * HDIM;
#pragma unroll
    for (int d = 0; d < 4; ++d)
      yrow[d * 16 + colc] = (_Float16)(O[d][v] * inv);
  }
}

// ---------------------------------------------------------------------------
// Kernel 3: output projection  Out = A @ Wo^T + bo (fp32 out).
// A: f16 [M=4096, K=1024] (async-staged) ; Wo: [N=1024, K=1024] fp32.
// ---------------------------------------------------------------------------
__global__ __launch_bounds__(256, 2)
void mha_out_gemm(const _Float16* __restrict__ A, const float* __restrict__ W,
                  const float* __restrict__ bias, float* __restrict__ Out) {
  __shared__ _Float16 As[TM][TK + 8];       // 16B-aligned rows for async b128
  __shared__ _Float16 Ws[TN][LDA];

  const int t = threadIdx.x;
  const int lane = t & 31;
  const int wave = t >> 5;
  const int wm = wave & 3;
  const int wn = wave >> 2;
  const int m0 = blockIdx.y * TM;
  const int n0 = blockIdx.x * TN;

  v8f acc[2][4];
#pragma unroll
  for (int i = 0; i < 2; ++i)
#pragma unroll
    for (int j = 0; j < 4; ++j) { v8f z = {}; acc[i][j] = z; }

  for (int k0 = 0; k0 < DIMSZ; k0 += TK) {
    // A tile: pure f16 copy -> async path (ASYNCcnt)
#pragma unroll
    for (int i = t; i < TM * 4; i += 256) {
      int r = i >> 2, c8 = (i & 3) * 8;
      async_copy16(A + (size_t)(m0 + r) * DIMSZ + k0 + c8, &As[r][c8]);
    }
    // W tile: fp32 -> f16 conversion through VGPRs
#pragma unroll
    for (int i = t; i < TN * 8; i += 256) {
      int r = i >> 3, c = (i & 7) * 4;
      float4 fw = *(const float4*)(W + (size_t)(n0 + r) * DIMSZ + k0 + c);
      Ws[r][c + 0] = (_Float16)fw.x; Ws[r][c + 1] = (_Float16)fw.y;
      Ws[r][c + 2] = (_Float16)fw.z; Ws[r][c + 3] = (_Float16)fw.w;
    }
    async_wait0();
    __syncthreads();

    v16h af[2], bf[4];
#pragma unroll
    for (int fm = 0; fm < 2; ++fm)
      af[fm] = load_afrag(&As[wm * 32 + fm * 16][0], TK + 8, lane);
#pragma unroll
    for (int fn = 0; fn < 4; ++fn)
      bf[fn] = load_bfrag_t(&Ws[wn * 64 + fn * 16][0], LDA, lane);
#pragma unroll
    for (int fm = 0; fm < 2; ++fm)
#pragma unroll
      for (int fn = 0; fn < 4; ++fn)
        acc[fm][fn] = wmma16(af[fm], bf[fn], acc[fm][fn]);
    __syncthreads();
  }

  const int colc = lane & 15;
  const int half8 = ((lane >> 4) & 1) * 8;
#pragma unroll
  for (int fm = 0; fm < 2; ++fm)
#pragma unroll
    for (int fn = 0; fn < 4; ++fn)
#pragma unroll
      for (int v = 0; v < 8; ++v) {
        int m = m0 + wm * 32 + fm * 16 + v + half8;
        int n = n0 + wn * 64 + fn * 16 + colc;
        Out[(size_t)m * DIMSZ + n] = acc[fm][fn][v] + bias[n];
      }
}

// ---------------------------------------------------------------------------
// Host launcher
// ---------------------------------------------------------------------------
extern "C" void kernel_launch(void* const* d_in, const int* in_sizes, int n_in,
                              void* d_out, int out_size, void* d_ws, size_t ws_size,
                              hipStream_t stream) {
  const float* q    = (const float*)d_in[0];
  const float* k    = (const float*)d_in[1];
  const float* v    = (const float*)d_in[2];
  const int*   mask = (const int*)  d_in[3];
  const float* Wq   = (const float*)d_in[4];
  const float* bq   = (const float*)d_in[5];
  const float* Wk   = (const float*)d_in[6];
  const float* bk   = (const float*)d_in[7];
  const float* Wv   = (const float*)d_in[8];
  const float* bv   = (const float*)d_in[9];
  const float* Wo   = (const float*)d_in[10];
  const float* bo   = (const float*)d_in[11];
  float* out = (float*)d_out;

  const size_t proj_elems = (size_t)MROWS * DIMSZ;   // 4096*1024
  _Float16* Qp = (_Float16*)d_ws;
  _Float16* Kp = Qp + proj_elems;
  _Float16* Vp = Kp + proj_elems;
  _Float16* Yp = Vp + proj_elems;

  dim3 gblk(256);
  dim3 ggrid(DIMSZ / TN, MROWS / TM);                // 8 x 32

  mha_proj_gemm<<<ggrid, gblk, 0, stream>>>(q, Wq, bq, Qp);
  mha_proj_gemm<<<ggrid, gblk, 0, stream>>>(k, Wk, bk, Kp);
  mha_proj_gemm<<<ggrid, gblk, 0, stream>>>(v, Wv, bv, Vp);

  dim3 agrid(SEQ / 64, BATCH * HEADS);               // 32 x 32
  mha_attn<<<agrid, dim3(128), 0, stream>>>(Qp, Kp, Vp, mask, Yp);

  mha_out_gemm<<<ggrid, gblk, 0, stream>>>(Yp, Wo, bo, out);
}
